// VisionTransformer_28673201668611
// MI455X (gfx1250) — matrix operand order repaired and checked by reading, not compiled, not run on hardware
//
#include <hip/hip_runtime.h>

#define DEV __device__ __forceinline__

typedef __bf16 bf16_t;
typedef __bf16 v16bf __attribute__((ext_vector_type(16)));
typedef __bf16 v8bf  __attribute__((ext_vector_type(8)));
typedef float  v8f   __attribute__((ext_vector_type(8)));
typedef unsigned int u32x4 __attribute__((ext_vector_type(4)));
typedef int v4i_gcc __attribute__((vector_size(16)));   // matches async builtin pointee

namespace vit {
constexpr int B = 32, S = 197, D = 768, H = 12, DK = 64, FF = 3072;
constexpr int L = 12, NP = 196, SP = 224, NC = 1000;
constexpr int MS = B * S;   // 6304 token rows
constexpr int MP = B * NP;  // 6272 patch rows
constexpr int RB = 40;      // LDS tile row stride (elems): 80B, 16B-aligned, bank-skewed
}

// CDNA5 async global->LDS path if the toolchain exposes the builtins.
#if __has_builtin(__builtin_amdgcn_global_load_async_to_lds_b128) && \
    __has_builtin(__builtin_amdgcn_s_wait_asynccnt)
#define VIT_ASYNC 1
#else
#define VIT_ASYNC 0
#endif

// ---------------- scalar helpers ----------------

DEV unsigned short f2bf(float f) {
  unsigned int u = __float_as_uint(f);
  u += 0x7FFFu + ((u >> 16) & 1u);      // round-to-nearest-even
  return (unsigned short)(u >> 16);
}

DEV float gelu_exact(float x) {
  return 0.5f * x * (1.f + erff(x * 0.70710678118654752f));
}

// block-wide reductions, blockDim.x == 256
DEV float block_sum(float v, float* red) {
  int t = threadIdx.x;
  red[t] = v; __syncthreads();
  for (int s = 128; s > 0; s >>= 1) { if (t < s) red[t] += red[t + s]; __syncthreads(); }
  float r = red[0]; __syncthreads();
  return r;
}
DEV float block_max(float v, float* red) {
  int t = threadIdx.x;
  red[t] = v; __syncthreads();
  for (int s = 128; s > 0; s >>= 1) { if (t < s) red[t] = fmaxf(red[t], red[t + s]); __syncthreads(); }
  float r = red[0]; __syncthreads();
  return r;
}

DEV v16bf shuf16(v8bf lo, v8bf hi) {
  return __builtin_shufflevector(lo, hi, 0, 1, 2, 3, 4, 5, 6, 7,
                                 8, 9, 10, 11, 12, 13, 14, 15);
}

// ---------------- tile staging ----------------
// One 128x32 bf16 tile = 512 x 16B chunks; 256 threads x 2 chunks each.
struct StageAddr {
  const unsigned short* g[2];
  int l[2];   // element offset into LDS tile (row stride RB)
};

DEV StageAddr stage_addr(const unsigned short* __restrict__ src, long ld, int kk,
                         int r0, int rmax) {
  using namespace vit;
  StageAddr s;
  const int t = threadIdx.x;
#pragma unroll
  for (int p = 0; p < 2; ++p) {
    int c = t + 256 * p;          // 0..511
    int row = c >> 2;             // 0..127
    int boff = (c & 3) << 3;      // 0,8,16,24 bf16 elems (16B chunks)
    int gr = r0 + row; if (gr >= rmax) gr = rmax - 1;   // clamp edge loads
    s.g[p] = src + (long)gr * ld + kk + boff;
    s.l[p] = row * RB + boff;
  }
  return s;
}

DEV void cp16_async(const unsigned short* g, unsigned short* l) {
#if VIT_ASYNC
  __builtin_amdgcn_global_load_async_to_lds_b128(
      (__attribute__((address_space(1))) v4i_gcc*)(void*)g,
      (__attribute__((address_space(3))) v4i_gcc*)(void*)l, 0, 0);
#else
  *(u32x4*)l = *(const u32x4*)g;
#endif
}

// ---------------- WMMA GEMM core ----------------
// C(128x128) = A(MxK bf16, row-major, lda) * B(NxK bf16, row-major, ldb)^T
// 256 threads = 8 waves, wave tile 32x64 (2x4 accumulators, 8 WMMA / K-step).
// Row-major LDS tiles; fragments are contiguous 16B chunks per lane:
//   A frag: [row][8*kh .. +8) ++ [row][16+8*kh .. +8)
//   B frag: [n  ][8*kh .. +8) ++ [n  ][16+8*kh .. +8)
// Double-buffered LDS; tile t+1 prefetched (async-to-LDS if available) during WMMAs.
__device__ void gemm_core(const unsigned short* __restrict__ A, long lda,
                          const unsigned short* __restrict__ Bw, long ldb, int K,
                          int m0, int n0, int M, int N, v8f acc[2][4]) {
  using namespace vit;
  __shared__ __attribute__((aligned(16))) unsigned short As[2][128][RB];
  __shared__ __attribute__((aligned(16))) unsigned short Bs[2][128][RB];

  const int lane = threadIdx.x & 31;
  const int wid  = threadIdx.x >> 5;
  const int wm   = wid >> 1, wn = wid & 1;   // wave grid 4(M) x 2(N)
  const int ln   = lane & 15, kh = lane >> 4;

#pragma unroll
  for (int i = 0; i < 2; ++i)
#pragma unroll
    for (int j = 0; j < 4; ++j)
#pragma unroll
      for (int t = 0; t < 8; ++t) acc[i][j][t] = 0.f;

  const int nT = K >> 5;

  // prologue: stage tile 0 into buffer 0
  {
    StageAddr sa = stage_addr(A,  lda, 0, m0, M);
    StageAddr sb = stage_addr(Bw, ldb, 0, n0, N);
#pragma unroll
    for (int p = 0; p < 2; ++p) cp16_async(sa.g[p], &As[0][0][0] + sa.l[p]);
#pragma unroll
    for (int p = 0; p < 2; ++p) cp16_async(sb.g[p], &Bs[0][0][0] + sb.l[p]);
#if VIT_ASYNC
    __builtin_amdgcn_s_wait_asynccnt(0);
#endif
    __syncthreads();
  }

  for (int t = 0; t < nT; ++t) {
    const int cur = t & 1;
    const bool pf = (t + 1 < nT);
    StageAddr sa, sb;
#if !VIT_ASYNC
    u32x4 pa[2], pb[2];
#endif
    if (pf) {
      sa = stage_addr(A,  lda, (t + 1) << 5, m0, M);
      sb = stage_addr(Bw, ldb, (t + 1) << 5, n0, N);
#if VIT_ASYNC
#pragma unroll
      for (int p = 0; p < 2; ++p) cp16_async(sa.g[p], &As[cur ^ 1][0][0] + sa.l[p]);
#pragma unroll
      for (int p = 0; p < 2; ++p) cp16_async(sb.g[p], &Bs[cur ^ 1][0][0] + sb.l[p]);
#else
#pragma unroll
      for (int p = 0; p < 2; ++p) pa[p] = *(const u32x4*)sa.g[p];
#pragma unroll
      for (int p = 0; p < 2; ++p) pb[p] = *(const u32x4*)sb.g[p];
#endif
    }

    // fragments + 8 WMMAs on current buffer
    v16bf af[2];
#pragma unroll
    for (int i = 0; i < 2; ++i) {
      const unsigned short* r = &As[cur][wm * 32 + i * 16 + ln][0];
      v8bf lo = *(const v8bf*)(r + 8 * kh);
      v8bf hi = *(const v8bf*)(r + 16 + 8 * kh);
      af[i] = shuf16(lo, hi);
    }
    v16bf bfr[4];
#pragma unroll
    for (int j = 0; j < 4; ++j) {
      const unsigned short* r = &Bs[cur][wn * 64 + j * 16 + ln][0];
      v8bf lo = *(const v8bf*)(r + 8 * kh);
      v8bf hi = *(const v8bf*)(r + 16 + 8 * kh);
      bfr[j] = shuf16(lo, hi);
    }
#pragma unroll
    for (int i = 0; i < 2; ++i)
#pragma unroll
      for (int j = 0; j < 4; ++j)
        acc[i][j] = __builtin_amdgcn_wmma_f32_16x16x32_bf16(
            false, af[i], false, bfr[j], (short)0, acc[i][j], false, false);

    if (pf) {
#if VIT_ASYNC
      __builtin_amdgcn_s_wait_asynccnt(0);
#else
#pragma unroll
      for (int p = 0; p < 2; ++p) *(u32x4*)(&As[cur ^ 1][0][0] + sa.l[p]) = pa[p];
#pragma unroll
      for (int p = 0; p < 2; ++p) *(u32x4*)(&Bs[cur ^ 1][0][0] + sb.l[p]) = pb[p];
#endif
    }
    __syncthreads();
  }
}

// C/D layout: lane holds (M = v + 8*(lane>>4), N = lane&15) for vgpr v
template <class F>
DEV void epilogue(v8f acc[2][4], int m0, int n0, int M, int N, F f) {
  const int lane = threadIdx.x & 31;
  const int wid  = threadIdx.x >> 5;
  const int wm   = wid >> 1, wn = wid & 1;
  const int mb = m0 + wm * 32 + ((lane >> 4) << 3);
  const int nb = n0 + wn * 64 + (lane & 15);
#pragma unroll
  for (int i = 0; i < 2; ++i)
#pragma unroll
    for (int j = 0; j < 4; ++j)
#pragma unroll
      for (int v = 0; v < 8; ++v) {
        int m = mb + i * 16 + v;
        int n = nb + j * 16;
        if (m < M && n < N) f(m, n, acc[i][j][v]);
      }
}

// ---------------- GEMM kernels ----------------

// f32 output; bias/residual presence are compile-time (in-place residual is fine)
template <bool BIAS, bool RES>
__global__ void __launch_bounds__(256)
k_gemm_f32out(const unsigned short* __restrict__ A, long lda,
              const unsigned short* __restrict__ W,
              const float* __restrict__ bias, const float* __restrict__ res,
              float* __restrict__ C, int M, int N, int K) {
  v8f acc[2][4];
  int m0 = blockIdx.x * 128, n0 = blockIdx.y * 128;
  gemm_core(A, lda, W, K, K, m0, n0, M, N, acc);
  epilogue(acc, m0, n0, M, N, [&](int m, int n, float v) {
    float o = v;
    if (BIAS) o += bias[n];
    if (RES)  o += res[(long)m * N + n];
    C[(long)m * N + n] = o;
  });
}

// bf16 output; optional bias / exact GELU
template <bool BIAS, bool GELU>
__global__ void __launch_bounds__(256)
k_gemm_bf16out(const unsigned short* __restrict__ A,
               const unsigned short* __restrict__ W,
               const float* __restrict__ bias, unsigned short* __restrict__ C,
               int M, int N, int K) {
  v8f acc[2][4];
  int m0 = blockIdx.x * 128, n0 = blockIdx.y * 128;
  gemm_core(A, K, W, K, K, m0, n0, M, N, acc);
  epilogue(acc, m0, n0, M, N, [&](int m, int n, float v) {
    float o = v;
    if (BIAS) o += bias[n];
    if (GELU) o = gelu_exact(o);
    C[(long)m * N + n] = f2bf(o);
  });
}

// V projection stored transposed (B,H,DK,SP) bf16, S padded to SP (pre-zeroed)
__global__ void __launch_bounds__(256)
k_gemm_vT(const unsigned short* __restrict__ A, const unsigned short* __restrict__ W,
          unsigned short* __restrict__ vt) {
  using namespace vit;
  v8f acc[2][4];
  int m0 = blockIdx.x * 128, n0 = blockIdx.y * 128;
  gemm_core(A, D, W, D, D, m0, n0, MS, D, acc);
  epilogue(acc, m0, n0, MS, D, [&](int m, int n, float v) {
    int b = m / S, s = m - b * S;
    int hh = n >> 6, d = n & 63;
    vt[(((long)b * H + hh) * DK + d) * SP + s] = f2bf(v);
  });
}

// scores = (Q K^T) / sqrt(DK), batched over B*H (blockIdx.z)
__global__ void __launch_bounds__(256)
k_attn_qk(const unsigned short* __restrict__ q, const unsigned short* __restrict__ k,
          float* __restrict__ sc) {
  using namespace vit;
  v8f acc[2][4];
  int z = blockIdx.z, b = z / H, hh = z - b * H;
  const unsigned short* aB = q + ((long)b * S) * D + hh * DK;
  const unsigned short* bB = k + ((long)b * S) * D + hh * DK;
  int m0 = blockIdx.x * 128, n0 = blockIdx.y * 128;
  gemm_core(aB, D, bB, D, DK, m0, n0, S, S, acc);
  float* out = sc + (long)z * S * S;
  epilogue(acc, m0, n0, S, S, [&](int m, int n, float v) {
    out[(long)m * S + n] = v * 0.125f;
  });
}

// ctx = attn * V, batched over B*H; K dim padded to SP with zeros
__global__ void __launch_bounds__(256)
k_attn_av(const unsigned short* __restrict__ at, const unsigned short* __restrict__ vt,
          unsigned short* __restrict__ ctx) {
  using namespace vit;
  v8f acc[2][4];
  int z = blockIdx.z, b = z / H, hh = z - b * H;
  const unsigned short* aB = at + (long)z * S * SP;
  const unsigned short* bB = vt + (long)z * DK * SP;
  int m0 = blockIdx.x * 128;
  gemm_core(aB, SP, bB, SP, SP, m0, 0, S, DK, acc);
  epilogue(acc, m0, 0, S, DK, [&](int m, int n, float v) {
    ctx[((long)b * S + m) * D + hh * DK + n] = f2bf(v);
  });
}

// ---------------- elementwise / reduction kernels ----------------

// f32 -> bf16 weight conversion (n multiple of 4)
__global__ void k_cvt_bf16(const float* __restrict__ src, unsigned short* __restrict__ dst,
                           long n) {
  long i = ((long)blockIdx.x * 256 + threadIdx.x) * 4;
  if (i < n) {
#pragma unroll
    for (int j = 0; j < 4; ++j) dst[i + j] = f2bf(src[i + j]);
  }
}

__global__ void k_im2col(const float* __restrict__ x, unsigned short* __restrict__ ap) {
  using namespace vit;
  long i = (long)blockIdx.x * blockDim.x + threadIdx.x;
  long total = (long)MP * D;
  if (i >= total) return;
  int kidx = (int)(i % D);
  long pidx = i / D;
  int b = (int)(pidx / NP);
  int pr = (int)(pidx - (long)b * NP);
  int py = pr / 14, px = pr - py * 14;
  int c = kidx >> 8;
  int r = kidx & 255;
  int ih = r >> 4, iw = r & 15;
  long xi = (((long)b * 3 + c) * 224 + (py * 16 + ih)) * 224 + (px * 16 + iw);
  ap[i] = f2bf(x[xi]);
}

__global__ void k_fill0(unsigned short* __restrict__ p, long n) {
  long i = (long)blockIdx.x * blockDim.x + threadIdx.x;
  if (i < n) p[i] = 0;
}

// h[row] = (s==0 ? cls : LN(pe_row; pn)) + pos[s]   (blockDim=256, one row/block)
__global__ void k_assemble(const float* __restrict__ pe, const float* __restrict__ cls,
                           const float* __restrict__ pos, const float* __restrict__ g,
                           const float* __restrict__ bb, float* __restrict__ h) {
  using namespace vit;
  __shared__ float red[256];
  int row = blockIdx.x;
  int b = row / S, s = row - b * S;
  int t = threadIdx.x;
  float o0, o1, o2;
  if (s == 0) {                               // uniform branch per block
    o0 = cls[t]; o1 = cls[t + 256]; o2 = cls[t + 512];
  } else {
    const float* xr = pe + ((long)b * NP + (s - 1)) * D;
    float v0 = xr[t], v1 = xr[t + 256], v2 = xr[t + 512];
    float mu = block_sum(v0 + v1 + v2, red) * (1.f / 768.f);
    float d0 = v0 - mu, d1 = v1 - mu, d2 = v2 - mu;
    float var = block_sum(d0 * d0 + d1 * d1 + d2 * d2, red) * (1.f / 768.f);
    float rs = rsqrtf(var + 1e-5f);
    o0 = d0 * rs * g[t]       + bb[t];
    o1 = d1 * rs * g[t + 256] + bb[t + 256];
    o2 = d2 * rs * g[t + 512] + bb[t + 512];
  }
  const float* pr = pos + (long)s * D;
  float* hr = h + (long)row * D;
  hr[t] = o0 + pr[t]; hr[t + 256] = o1 + pr[t + 256]; hr[t + 512] = o2 + pr[t + 512];
}

// LayerNorm(768) -> bf16 out (blockDim=256, one row/block)
__global__ void k_ln(const float* __restrict__ in, const float* __restrict__ g,
                     const float* __restrict__ bb, unsigned short* __restrict__ out) {
  __shared__ float red[256];
  long row = blockIdx.x;
  const float* xr = in + row * 768;
  int t = threadIdx.x;
  float v0 = xr[t], v1 = xr[t + 256], v2 = xr[t + 512];
  float mu = block_sum(v0 + v1 + v2, red) * (1.f / 768.f);
  float d0 = v0 - mu, d1 = v1 - mu, d2 = v2 - mu;
  float var = block_sum(d0 * d0 + d1 * d1 + d2 * d2, red) * (1.f / 768.f);
  float rs = rsqrtf(var + 1e-5f);
  unsigned short* orow = out + row * 768;
  orow[t]       = f2bf(d0 * rs * g[t]       + bb[t]);
  orow[t + 256] = f2bf(d1 * rs * g[t + 256] + bb[t + 256]);
  orow[t + 512] = f2bf(d2 * rs * g[t + 512] + bb[t + 512]);
}

// softmax over 197 score cols -> bf16 attn row padded to 224 (blockDim=256)
__global__ void k_softmax_attn(const float* __restrict__ sc, unsigned short* __restrict__ at) {
  using namespace vit;
  __shared__ float red[256];
  long row = blockIdx.x;     // 0 .. B*H*S-1
  int t = threadIdx.x;
  float v = (t < S) ? sc[row * S + t] : -3.4e38f;
  float mx = block_max(v, red);
  float e = (t < S) ? __expf(v - mx) : 0.f;
  float sum = block_sum(e, red);
  if (t < SP) at[row * SP + t] = f2bf((t < S) ? e / sum : 0.f);
}

// softmax over 1000 logits -> f32 output (blockDim=256, one batch row/block)
__global__ void k_softmax_out(const float* __restrict__ logits, float* __restrict__ out) {
  using namespace vit;
  __shared__ float red[256];
  int b = blockIdx.x, t = threadIdx.x;
  float v[4];
  float mx = -3.4e38f;
#pragma unroll
  for (int j = 0; j < 4; ++j) {
    int c = t + j * 256;
    v[j] = (c < NC) ? logits[(long)b * NC + c] : -3.4e38f;
    mx = fmaxf(mx, v[j]);
  }
  mx = block_max(mx, red);
  float s = 0.f;
#pragma unroll
  for (int j = 0; j < 4; ++j) {
    int c = t + j * 256;
    v[j] = (c < NC) ? __expf(v[j] - mx) : 0.f;
    s += v[j];
  }
  s = block_sum(s, red);
#pragma unroll
  for (int j = 0; j < 4; ++j) {
    int c = t + j * 256;
    if (c < NC) out[(long)b * NC + c] = v[j] / s;
  }
}

// ---------------- orchestration ----------------

extern "C" void kernel_launch(void* const* d_in, const int* in_sizes, int n_in,
                              void* d_out, int out_size, void* d_ws, size_t ws_size,
                              hipStream_t stream) {
  using namespace vit;
  (void)in_sizes; (void)n_in; (void)out_size; (void)ws_size;

  const float* x      = (const float*)d_in[0];
  const float* conv_w = (const float*)d_in[1];
  const float* conv_b = (const float*)d_in[2];
  const float* pn_g   = (const float*)d_in[3];
  const float* pn_b   = (const float*)d_in[4];
  const float* cls    = (const float*)d_in[5];
  const float* pos    = (const float*)d_in[6];
  const float* ln1_g  = (const float*)d_in[7];
  const float* ln1_b  = (const float*)d_in[8];
  const float* Wq     = (const float*)d_in[9];
  const float* Wk     = (const float*)d_in[10];
  const float* Wv     = (const float*)d_in[11];
  const float* Wo     = (const float*)d_in[12];
  const float* ln2_g  = (const float*)d_in[13];
  const float* ln2_b  = (const float*)d_in[14];
  const float* fc1_w  = (const float*)d_in[15];
  const float* fc1_b  = (const float*)d_in[16];
  const float* fc2_w  = (const float*)d_in[17];
  const float* fc2_b  = (const float*)d_in[18];
  const float* lnf_g  = (const float*)d_in[19];
  const float* lnf_b  = (const float*)d_in[20];
  const float* head_w = (const float*)d_in[21];
  const float* head_b = (const float*)d_in[22];
  float* out = (float*)d_out;

  // workspace carve-up
  char* base = (char*)d_ws;
  size_t off = 0;
  auto alloc = [&](size_t bytes) -> void* {
    void* p = base + off;
    off += (bytes + 255) & ~(size_t)255;
    return p;
  };
  unsigned short* apatch = (unsigned short*)alloc((size_t)MP * D * 2);
  float*          pe     = (float*)         alloc((size_t)MP * D * 4);
  float*          h      = (float*)         alloc((size_t)MS * D * 4);
  unsigned short* hn     = (unsigned short*)alloc((size_t)MS * D * 2);
  unsigned short* q      = (unsigned short*)alloc((size_t)MS * D * 2);
  unsigned short* kbuf   = (unsigned short*)alloc((size_t)MS * D * 2);
  unsigned short* vt     = (unsigned short*)alloc((size_t)B * H * DK * SP * 2);
  float*          scores = (float*)         alloc((size_t)B * H * S * S * 4);
  unsigned short* attn   = (unsigned short*)alloc((size_t)B * H * S * SP * 2);
  unsigned short* ctx    = (unsigned short*)alloc((size_t)MS * D * 2);
  unsigned short* ff     = (unsigned short*)alloc((size_t)MS * FF * 2);
  float*          logits = (float*)         alloc((size_t)B * NC * 4);
  unsigned short* wbf    = (unsigned short*)alloc((size_t)FF * D * 2);  // weight staging

  const dim3 blk(256);
  const dim3 gTok((MS + 127) / 128, (D + 127) / 128);   // 50 x 6
  const dim3 gFF((MS + 127) / 128, (FF + 127) / 128);   // 50 x 24

  auto cvt = [&](const float* w, long n) {
    k_cvt_bf16<<<dim3((unsigned)((n + 1023) / 1024)), 256, 0, stream>>>(w, wbf, n);
  };

  // ---- patch embedding ----
  {
    long n = (long)MP * D;
    k_im2col<<<dim3((unsigned)((n + 255) / 256)), 256, 0, stream>>>(x, apatch);
  }
  cvt(conv_w, (long)D * D);
  k_gemm_f32out<true, false><<<dim3((MP + 127) / 128, (D + 127) / 128), blk, 0, stream>>>(
      apatch, D, wbf, conv_b, nullptr, pe, MP, D, D);
  k_assemble<<<dim3(MS), 256, 0, stream>>>(pe, cls, pos, pn_g, pn_b, h);

  // ---- encoder layers ----
  for (int i = 0; i < L; ++i) {
    const float* wq = Wq + (size_t)i * D * D;
    const float* wk = Wk + (size_t)i * D * D;
    const float* wv = Wv + (size_t)i * D * D;
    const float* wo = Wo + (size_t)i * D * D;
    const float* w1 = fc1_w + (size_t)i * FF * D;
    const float* b1 = fc1_b + (size_t)i * FF;
    const float* w2 = fc2_w + (size_t)i * D * FF;
    const float* b2 = fc2_b + (size_t)i * D;

    k_ln<<<dim3(MS), 256, 0, stream>>>(h, ln1_g + (size_t)i * D, ln1_b + (size_t)i * D, hn);

    cvt(wq, (long)D * D);
    k_gemm_bf16out<false, false><<<gTok, blk, 0, stream>>>(hn, wbf, nullptr, q, MS, D, D);
    cvt(wk, (long)D * D);
    k_gemm_bf16out<false, false><<<gTok, blk, 0, stream>>>(hn, wbf, nullptr, kbuf, MS, D, D);
    {
      long n = (long)B * H * DK * SP;
      k_fill0<<<dim3((unsigned)((n + 255) / 256)), 256, 0, stream>>>(vt, n);
    }
    cvt(wv, (long)D * D);
    k_gemm_vT<<<gTok, blk, 0, stream>>>(hn, wbf, vt);

    k_attn_qk<<<dim3((S + 127) / 128, (S + 127) / 128, B * H), blk, 0, stream>>>(q, kbuf, scores);
    k_softmax_attn<<<dim3(B * H * S), 256, 0, stream>>>(scores, attn);
    k_attn_av<<<dim3((S + 127) / 128, 1, B * H), blk, 0, stream>>>(attn, vt, ctx);

    cvt(wo, (long)D * D);
    k_gemm_f32out<false, true><<<gTok, blk, 0, stream>>>(ctx, D, wbf, nullptr, h, h, MS, D, D);

    k_ln<<<dim3(MS), 256, 0, stream>>>(h, ln2_g + (size_t)i * D, ln2_b + (size_t)i * D, hn);
    cvt(w1, (long)FF * D);
    k_gemm_bf16out<true, true><<<gFF, blk, 0, stream>>>(hn, wbf, b1, ff, MS, FF, D);
    cvt(w2, (long)D * FF);
    k_gemm_f32out<true, true><<<gTok, blk, 0, stream>>>(ff, FF, wbf, b2, h, h, MS, D, FF);
  }

  // ---- head ----
  k_ln<<<dim3(MS), 256, 0, stream>>>(h, lnf_g, lnf_b, hn);
  cvt(head_w, (long)NC * D);
  // rows b -> hn[b*S + 0] via lda = S*D
  k_gemm_f32out<true, false><<<dim3(1, (NC + 127) / 128), blk, 0, stream>>>(
      hn, (long)S * D, wbf, head_b, nullptr, logits, B, NC, D);
  k_softmax_out<<<dim3(B), 256, 0, stream>>>(logits, out);
}
